// LSTM_16527034155504
// MI455X (gfx1250) — matrix operand order repaired and checked by reading, not compiled, not run on hardware
//
#include <hip/hip_runtime.h>
#include <stddef.h>

// ---------------------------------------------------------------------------
// CDNA5 (gfx1250) LSTM:  xz = x@Wx+b  (bulk WMMA GEMM, whole GPU)
//                        recurrence   (persistent 1-WGP kernel, h in LDS)
//                        head         (WMMA + softmax)
// Matrix math: v_wmma_f32_16x16x32_bf16 (wave32 fragments).
// Block copies LDS<->global: global_{load,store}_async_*_lds_b128 + ASYNCcnt.
// ---------------------------------------------------------------------------

#define USE_GFX1250_ASYNC_COPY 1

typedef __attribute__((ext_vector_type(16))) __bf16 v16bf;
typedef __attribute__((ext_vector_type(8)))  float  v8f;

// bf16 bit helpers (bf16 pattern == fp32 >> 16)
static __device__ __forceinline__ unsigned short f2bf_bits(float f) {
    return __builtin_bit_cast(unsigned short, (__bf16)f);
}
static __device__ __forceinline__ float bf2f(unsigned short u) {
    unsigned int x = ((unsigned int)u) << 16;
    return __builtin_bit_cast(float, x);
}
static __device__ __forceinline__ unsigned int pack2(float a, float b) {
    return (unsigned int)f2bf_bits(a) | ((unsigned int)f2bf_bits(b) << 16);
}
static __device__ __forceinline__ float sigmoidf_fast(float x) {
    return 1.0f / (1.0f + __expf(-x));
}

#if USE_GFX1250_ASYNC_COPY
// ASYNCcnt-tracked 16B block copies between LDS and global memory (gfx1250).
// lds_off is a byte offset from the wave's LDS base (dynamic-LDS starts at 0
// in kernels with no static __shared__ allocations).
static __device__ __forceinline__ void async_lds_to_global_b128(void* gaddr, unsigned lds_off) {
    asm volatile("global_store_async_from_lds_b128 %0, %1, off"
                 :: "v"(gaddr), "v"(lds_off) : "memory");
}
static __device__ __forceinline__ void async_global_to_lds_b128(unsigned lds_off, const void* gaddr) {
    asm volatile("global_load_async_to_lds_b128 %0, %1, off"
                 :: "v"(lds_off), "v"(gaddr) : "memory");
}
static __device__ __forceinline__ void wait_asynccnt0() {
    asm volatile("s_wait_asynccnt 0x0" ::: "memory");
}
#endif

// ---------------------------------------------------------------------------
// Pack a [K, 4*gateW] (4 gate matrices concatenated on N) f32 weight into
// bf16 WMMA B-fragment order:
//   packed[((kt*NT + nt)*32 + lane)*16 + e] = W[kt*32 + (lane/16)*8 + e + (e>=8)*8][nt*16 + lane%16]
// (B 32x16 layout: K = (lane/16)*8 + e + (e>=8)*8, col = lane%16)
// ---------------------------------------------------------------------------
__global__ void pack_b_kernel(__bf16* __restrict__ dst,
                              const float* __restrict__ p0, const float* __restrict__ p1,
                              const float* __restrict__ p2, const float* __restrict__ p3,
                              int NT, int gateW, int total)
{
    int idx = blockIdx.x * 256 + threadIdx.x;
    if (idx >= total) return;
    int e    = idx & 15;
    int lane = (idx >> 4) & 31;
    int tile = idx >> 9;
    int nt   = tile % NT;
    int kt   = tile / NT;
    int row  = kt * 32 + (lane >> 4) * 8 + e + ((e >> 3) << 3);
    int col  = nt * 16 + (lane & 15);
    int gate = col / gateW;
    int c2   = col % gateW;
    const float* p = (gate == 0) ? p0 : (gate == 1) ? p1 : (gate == 2) ? p2 : p3;
    dst[idx] = (__bf16)p[(size_t)row * gateW + c2];
}

// ---------------------------------------------------------------------------
// Phase 1: xz[t,b,:] = x[b,t,:] @ Wx + bias      (M = T*B, K = 128, N = 1024)
// One block per 16-row M-tile (blockIdx.x = t*8 + mt), 8 waves, each wave
// owns 8 N-tiles. A fragments are built once per block into LDS (4 KB) and
// shared by all waves via ds_load. Output stored bf16 in D-fragment order:
//   xzP[(((t*8+mt)*64 + nt)*32 + lane)*8 + r]
// ---------------------------------------------------------------------------
__global__ __launch_bounds__(256) void lstm_xz_kernel(
    const float* __restrict__ x, const __bf16* __restrict__ WxP,
    const float* __restrict__ b0, const float* __restrict__ b1,
    const float* __restrict__ b2, const float* __restrict__ b3,
    __bf16* __restrict__ xzP)
{
    __shared__ __bf16 aS[2048];       // 4 K-tiles of A fragments, [kt][lane][e]

    const int bx   = blockIdx.x;      // t*8 + mt
    const int t    = bx >> 3;
    const int mt   = bx & 7;
    const int tid  = threadIdx.x;
    const int wave = tid >> 5;
    const int lane = tid & 31;
    const int n    = lane & 15;

    // Cooperative A-fragment build (each thread fills 8 bf16 slots).
    // A layout (16x32 bf16): row = lane%16, K = (lane/16)*8 + e + (e>=8)*8.
    for (int idx = tid; idx < 2048; idx += 256) {
        const int kt2 = idx >> 9;
        const int l2  = (idx >> 4) & 31;
        const int e   = idx & 15;
        const int K   = (l2 >> 4) * 8 + e + ((e >> 3) << 3);
        const int br  = mt * 16 + (l2 & 15);
        aS[idx] = (__bf16)x[((size_t)br * 2048 + t) * 128 + kt2 * 32 + K];
    }
    __syncthreads();

    v16bf A[4];
#pragma unroll
    for (int kt = 0; kt < 4; ++kt)
        A[kt] = *(const v16bf*)(aS + (kt * 32 + lane) * 16);

#pragma unroll
    for (int ni = 0; ni < 8; ++ni) {
        const int nt   = wave * 8 + ni;
        const int col  = nt * 16 + n;
        const int gate = col >> 8;
        const int bc   = col & 255;
        const float* bp = (gate == 0) ? b0 : (gate == 1) ? b1 : (gate == 2) ? b2 : b3;
        const float bb = bp[bc];
        v8f acc;
#pragma unroll
        for (int r = 0; r < 8; ++r) acc[r] = bb;
#pragma unroll
        for (int kt = 0; kt < 4; ++kt) {
            v16bf B = *(const v16bf*)(WxP + ((size_t)(kt * 64 + nt) * 32 + lane) * 16);
            acc = __builtin_amdgcn_wmma_f32_16x16x32_bf16(false, A[kt], false, B,
                                                          (short)0, acc, false, false);
        }
        uint4 st;
        st.x = pack2(acc[0], acc[1]);
        st.y = pack2(acc[2], acc[3]);
        st.z = pack2(acc[4], acc[5]);
        st.w = pack2(acc[6], acc[7]);
        *(uint4*)(xzP + ((((size_t)bx) * 64 + nt) * 32 + lane) * 8) = st;
    }
}

// ---------------------------------------------------------------------------
// Phase 2: sequential recurrence. ONE workgroup, 1024 threads = 32 waves on
// one WGP. LDS: h packed as bf16 A-fragments (64 KB) + c as f32 (128 KB).
// Wave w: jt = w%16 (h-column tile), mhalf = w/16 (row-tile half).
// Each wave computes, for its 4 row-tiles, the 4 gate tiles of its columns
// (nt = q*16 + jt), so the LSTM cell update is fully wave-local.
// ---------------------------------------------------------------------------
__global__ __launch_bounds__(1024) void lstm_recur_kernel(
    const __bf16* __restrict__ xzP, const __bf16* __restrict__ WhP,
    __bf16* __restrict__ hP)
{
    extern __shared__ char smem[];
    __bf16* hA = (__bf16*)smem;              // 32768 bf16 = 64 KB (A-fragment order)
    float*  cB = (float*)(smem + 65536);     // 32768 f32  = 128 KB, cB[b*256 + j]

    const int tid   = threadIdx.x;
    const int wave  = tid >> 5;
    const int lane  = tid & 31;
    const int n     = lane & 15;
    const int g     = lane >> 4;
    const int jt    = wave & 15;
    const int mhalf = wave >> 4;

    // zero-init h0, c0
    for (int i = tid; i < 32768; i += 1024) { ((unsigned short*)hA)[i] = 0; cB[i] = 0.0f; }
    __syncthreads();

    // Per-lane h-column and its A-fragment write slot (row-independent parts)
    const int j   = jt * 16 + n;            // hidden column 0..255
    const int ktH = j >> 5;                 // A K-tile of this column
    const int K   = j & 31;
    const int g2  = (K >> 3) & 1;           // A layout inversion
    const int e2  = (K & 7) + ((K >> 4) << 3);

    unsigned short pend[4][8];              // pending h (bf16 bits) per row-tile

    for (int t = 0; t < 2048; ++t) {
#pragma unroll 1
        for (int ml = 0; ml < 4; ++ml) {
            const int mt = mhalf * 4 + ml;

            // accumulators initialized from precomputed xz (bf16 fragments)
            v8f acc[4];
#pragma unroll
            for (int q = 0; q < 4; ++q) {
                const int nt = q * 16 + jt;
                const size_t off = ((((size_t)t * 8 + mt) * 64 + nt) * 32 + lane) * 8;
                uint4 u = *(const uint4*)(xzP + off);
                if (t + 1 < 2048)   // warm L2/GL1 for the next timestep
                    __builtin_prefetch(xzP + off + (size_t)8 * 64 * 32 * 8, 0, 3);
                unsigned int wd[4] = {u.x, u.y, u.z, u.w};
#pragma unroll
                for (int r = 0; r < 8; ++r)
                    acc[q][r] = bf2f((unsigned short)((r & 1) ? (wd[r >> 1] >> 16)
                                                              : (wd[r >> 1] & 0xffffu)));
            }

            // z += h @ Wh : 8 K-tiles, A from LDS, B streamed from L2
#pragma unroll
            for (int kt = 0; kt < 8; ++kt) {
                v16bf a = *(const v16bf*)(hA + ((size_t)(mt * 8 + kt) * 32 + lane) * 16);
#pragma unroll
                for (int q = 0; q < 4; ++q) {
                    const int nt = q * 16 + jt;
                    v16bf b = *(const v16bf*)(WhP + ((size_t)(kt * 64 + nt) * 32 + lane) * 16);
                    acc[q] = __builtin_amdgcn_wmma_f32_16x16x32_bf16(false, a, false, b,
                                                                     (short)0, acc[q], false, false);
                }
            }

            // LSTM cell update (wave owns all 4 gates of column j, rows of mt)
#pragma unroll
            for (int r = 0; r < 8; ++r) {
                const int brow = mt * 16 + g * 8 + r;     // batch row
                float zg = acc[0][r], zi = acc[1][r], zf = acc[2][r], zo = acc[3][r];
                float co = cB[brow * 256 + j];
                float gg = tanhf(zg);
                float ii = sigmoidf_fast(zi);
                float ff = sigmoidf_fast(zf);
                float oo = sigmoidf_fast(zo);
                float cn = gg * ii + co * ff;
                cB[brow * 256 + j] = cn;                  // exclusive ownership
                pend[ml][r] = f2bf_bits(tanhf(cn) * oo);
            }
        }

        __syncthreads();   // all waves done READING hA(t-1)

        // publish h(t) into the A-fragment LDS layout
#pragma unroll
        for (int ml = 0; ml < 4; ++ml) {
            const int mt = mhalf * 4 + ml;
#pragma unroll
            for (int r = 0; r < 8; ++r) {
                const int mrow = g * 8 + r;               // row within tile
                const int Lp   = g2 * 16 + mrow;
                ((unsigned short*)hA)[((size_t)(mt * 8 + ktH) * 32 + Lp) * 16 + e2] = pend[ml][r];
            }
        }

        __syncthreads();   // h(t) visible before next step's WMMA reads
    }

    // export final h (already in A-fragment order) for the head kernel:
    // 64 KB LDS -> global as ASYNCcnt-tracked 16B block transfers.
#if USE_GFX1250_ASYNC_COPY
#pragma unroll
    for (int i = 0; i < 4; ++i) {
        const unsigned c = (unsigned)(i * 1024 + tid);    // 4096 chunks of 16B
        async_lds_to_global_b128((char*)hP + (size_t)c * 16, c * 16u);
    }
    wait_asynccnt0();
#else
    for (int i = tid; i < 32768; i += 1024)
        ((unsigned short*)hP)[i] = ((unsigned short*)hA)[i];
#endif
}

// ---------------------------------------------------------------------------
// Phase 3: logits = h @ W_ph + b_p  (128x256x128), then row softmax.
// One workgroup, 8 waves; wave w owns N-tile w; loops over 8 M-tiles.
// hP is async-staged into LDS once (it is read by all 8 waves).
// LDS: [0,64K) staged h fragments, [64K,128K) f32 logits.
// ---------------------------------------------------------------------------
__global__ __launch_bounds__(256) void lstm_head_kernel(
    const __bf16* __restrict__ hP, const __bf16* __restrict__ WpP,
    const float* __restrict__ bp, float* __restrict__ out)
{
    extern __shared__ char smem[];
    __bf16* hS = (__bf16*)smem;                 // 64 KB staged h fragments
    float*  lg = (float*)(smem + 65536);        // 128*128 f32 logits

    const int tid  = threadIdx.x;
    const int wave = tid >> 5;
    const int lane = tid & 31;
    const int n    = lane & 15;
    const int g    = lane >> 4;
    const int nt   = wave;

    // stage hP into LDS (4096 chunks of 16B over 256 threads)
#if USE_GFX1250_ASYNC_COPY
#pragma unroll
    for (int i = 0; i < 16; ++i) {
        const unsigned c = (unsigned)(i * 256 + tid);
        async_global_to_lds_b128(c * 16u, (const char*)hP + (size_t)c * 16);
    }
    wait_asynccnt0();
#else
    for (int i = 0; i < 16; ++i) {
        const int c = i * 256 + tid;
        *(uint4*)((char*)smem + (size_t)c * 16) = *(const uint4*)((const char*)hP + (size_t)c * 16);
    }
#endif
    __syncthreads();

    const float bb = bp[nt * 16 + n];
#pragma unroll 1
    for (int mt = 0; mt < 8; ++mt) {
        v8f acc;
#pragma unroll
        for (int r = 0; r < 8; ++r) acc[r] = bb;
#pragma unroll
        for (int kt = 0; kt < 8; ++kt) {
            v16bf a = *(const v16bf*)(hS  + ((size_t)(mt * 8 + kt) * 32 + lane) * 16);
            v16bf b = *(const v16bf*)(WpP + ((size_t)(kt * 8 + nt) * 32 + lane) * 16);
            acc = __builtin_amdgcn_wmma_f32_16x16x32_bf16(false, a, false, b,
                                                          (short)0, acc, false, false);
        }
#pragma unroll
        for (int r = 0; r < 8; ++r)
            lg[(mt * 16 + g * 8 + r) * 128 + nt * 16 + n] = acc[r];
    }
    __syncthreads();

    // softmax, one thread per row; recompute exp to avoid per-thread arrays
    if (tid < 128) {
        const int row = tid;
        float mx = -3.402823466e+38f;
        for (int c = 0; c < 128; ++c) mx = fmaxf(mx, lg[row * 128 + c]);
        float s = 0.0f;
        for (int c = 0; c < 128; ++c) s += __expf(lg[row * 128 + c] - mx);
        const float inv = 1.0f / s;
        for (int c = 0; c < 128; ++c) out[row * 128 + c] = __expf(lg[row * 128 + c] - mx) * inv;
    }
}

// ---------------------------------------------------------------------------
// Host launcher. Workspace layout (bytes):
//   [0,       262144)  WxP  bf16  (128 x1024 packed)
//   [262144,  786432)  WhP  bf16  (256 x1024 packed)
//   [786432,  851968)  WpP  bf16  (256 x 128 packed)
//   [851968,  917504)  hP   bf16  (final h, A-fragment order)
//   [917504,  +512MB)  xzP  bf16  (T x 8mt x 64nt x 32lane x 8r)
// ---------------------------------------------------------------------------
extern "C" void kernel_launch(void* const* d_in, const int* in_sizes, int n_in,
                              void* d_out, int out_size, void* d_ws, size_t ws_size,
                              hipStream_t stream)
{
    (void)in_sizes; (void)n_in; (void)out_size; (void)ws_size;
    const float* x   = (const float*)d_in[0];
    const float* Wgx = (const float*)d_in[1];
    const float* Wgh = (const float*)d_in[2];
    const float* bg  = (const float*)d_in[3];
    const float* Wix = (const float*)d_in[4];
    const float* Wih = (const float*)d_in[5];
    const float* bi  = (const float*)d_in[6];
    const float* Wfx = (const float*)d_in[7];
    const float* Wfh = (const float*)d_in[8];
    const float* bf  = (const float*)d_in[9];
    const float* Wox = (const float*)d_in[10];
    const float* Woh = (const float*)d_in[11];
    const float* bo  = (const float*)d_in[12];
    const float* Wph = (const float*)d_in[13];
    const float* bp  = (const float*)d_in[14];

    char* w = (char*)d_ws;
    __bf16* WxP = (__bf16*)(w + 0);
    __bf16* WhP = (__bf16*)(w + 262144);
    __bf16* WpP = (__bf16*)(w + 786432);
    __bf16* hP  = (__bf16*)(w + 851968);
    __bf16* xzP = (__bf16*)(w + 917504);

    // pack weights into WMMA B-fragment order
    pack_b_kernel<<<512,  256, 0, stream>>>(WxP, Wgx, Wix, Wfx, Wox, 64, 256, 131072);
    pack_b_kernel<<<1024, 256, 0, stream>>>(WhP, Wgh, Wih, Wfh, Woh, 64, 256, 262144);
    pack_b_kernel<<<128,  256, 0, stream>>>(WpP, Wph, Wph, Wph, Wph,  8, 128,  32768);

    // bulk input projection across the whole GPU
    lstm_xz_kernel<<<16384, 256, 0, stream>>>(x, WxP, bg, bi, bf, bo, xzP);

    // sequential recurrence: one persistent WGP-resident workgroup, 192 KB LDS
    lstm_recur_kernel<<<1, 1024, 196608, stream>>>(xzP, WhP, hP);

    // head GEMM + softmax (128 KB dynamic LDS: staged h + logits)
    lstm_head_kernel<<<1, 256, 131072, stream>>>(hP, WpP, bp, (float*)d_out);
}
